// SimpleSparseConvModel_20564303414101
// MI455X (gfx1250) — matrix operand order repaired and checked by reading, not compiled, not run on hardware
//
#include <hip/hip_runtime.h>
#include <hip/hip_bf16.h>

typedef __attribute__((ext_vector_type(16))) __bf16 v16bf;
typedef __attribute__((ext_vector_type(8)))  float  v8f;

#define C_IN   32
#define C_OUT  64
#define KVOL   27
#define BN_EPS 1e-5f

// ---------- helpers ----------
__device__ __forceinline__ unsigned short f2bf(float f) {
    unsigned u = __builtin_bit_cast(unsigned, f);
    unsigned r = u + 0x7FFFu + ((u >> 16) & 1u);   // round-to-nearest-even
    return (unsigned short)(r >> 16);
}

union Frag { uint4 q[2]; v16bf v; };

// ---------- kernel 1a: x (f32) -> bf16 ----------
__global__ void cvt_x_bf16(const float* __restrict__ x,
                           unsigned short* __restrict__ xb, int total4) {
    int i = blockIdx.x * blockDim.x + threadIdx.x;
    if (i < total4) {
        const float4 v = reinterpret_cast<const float4*>(x)[i];
        ushort4 o;
        o.x = f2bf(v.x); o.y = f2bf(v.y); o.z = f2bf(v.z); o.w = f2bf(v.w);
        reinterpret_cast<ushort4*>(xb)[i] = o;
    }
}

// ---------- kernel 1b: pack W[k][c][d] into per-lane B fragments ----------
// wp[((k*4 + t)*32 + lane)*16 + e] = bf16( W[k][ (e&7) + 8*(lane>>4) + 16*(e>>3) ][ 16*t + (lane&15) ] )
__global__ void pack_w(const float* __restrict__ W, unsigned short* __restrict__ wp) {
    int i = blockIdx.x * blockDim.x + threadIdx.x;          // 27*4*32*16 = 55296
    if (i >= KVOL * 4 * 32 * 16) return;
    int e    = i & 15;
    int lane = (i >> 4) & 31;
    int t    = (i >> 9) & 3;
    int k    = i >> 11;
    int c    = (e & 7) + 8 * (lane >> 4) + 16 * (e >> 3);
    int d    = 16 * t + (lane & 15);
    wp[i] = f2bf(W[(k * C_IN + c) * C_OUT + d]);
}

// ---------- kernel 2: gather + bf16 WMMA GEMM + per-block BN partials ----------
// 256 threads = 8 waves; wave -> 32 rows x 64 cols (2 M-tiles x 4 N-tiles); block -> 256 rows.
__global__ void __launch_bounds__(256)
sparse_conv_gemm(const unsigned short* __restrict__ xb,
                 const unsigned short* __restrict__ wp,
                 const int* __restrict__ idx,
                 float* __restrict__ y,            // d_out, pre-BN
                 float* __restrict__ partial) {    // [nblocks][128] sums|sumsq
    __shared__ float sS[8 * 128];
    __shared__ float sQ[8 * 128];

    const int wave = threadIdx.x >> 5;
    const int lane = threadIdx.x & 31;
    const int r    = lane & 15;        // row within M-tile (A) / col within N-tile (B,D)
    const int h    = lane >> 4;        // lane half
    const int rowbase = blockIdx.x * 256 + wave * 32;

    v8f acc[8];                        // [t] = rows 0..15, [4+t] = rows 16..31
#pragma unroll
    for (int t = 0; t < 8; ++t) acc[t] = (v8f){0.f,0.f,0.f,0.f,0.f,0.f,0.f,0.f};

    const int* idxrow0 = idx + (size_t)(rowbase + r) * KVOL;
    const int* idxrow1 = idx + (size_t)(rowbase + 16 + r) * KVOL;
    const unsigned short* wlane = wp + (size_t)lane * 16;

#pragma unroll 1
    for (int k = 0; k < KVOL; ++k) {
        // two gathered A fragments; per-lane K chunks [8h..8h+7] and [16+8h..16+8h+7]
        const unsigned short* xr0 = xb + (size_t)idxrow0[k] * C_IN;
        const unsigned short* xr1 = xb + (size_t)idxrow1[k] * C_IN;
        Frag a0, a1;
        a0.q[0] = *reinterpret_cast<const uint4*>(xr0 + 8 * h);
        a0.q[1] = *reinterpret_cast<const uint4*>(xr0 + 16 + 8 * h);
        a1.q[0] = *reinterpret_cast<const uint4*>(xr1 + 8 * h);
        a1.q[1] = *reinterpret_cast<const uint4*>(xr1 + 16 + 8 * h);

#pragma unroll
        for (int t = 0; t < 4; ++t) {
            Frag b;
            const uint4* bp =
                reinterpret_cast<const uint4*>(wlane + (size_t)(k * 4 + t) * 512);
            b.q[0] = bp[0];
            b.q[1] = bp[1];
            acc[t] = __builtin_amdgcn_wmma_f32_16x16x32_bf16(
                false, a0.v, false, b.v, (short)0, acc[t], false, false);
            acc[4 + t] = __builtin_amdgcn_wmma_f32_16x16x32_bf16(
                false, a1.v, false, b.v, (short)0, acc[4 + t], false, false);
        }
    }

    // store pre-BN y and accumulate per-lane column sums
#pragma unroll
    for (int t = 0; t < 4; ++t) {
        float s = 0.f, q = 0.f;
#pragma unroll
        for (int i = 0; i < 8; ++i) {
            const float v0 = acc[t][i];
            const float v1 = acc[4 + t][i];
            y[(size_t)(rowbase + i + 8 * h) * C_OUT + 16 * t + r]      = v0;
            y[(size_t)(rowbase + 16 + i + 8 * h) * C_OUT + 16 * t + r] = v1;
            s += v0 + v1;
            q += v0 * v0 + v1 * v1;
        }
        sS[wave * 128 + t * 32 + lane] = s;
        sQ[wave * 128 + t * 32 + lane] = q;
    }
    __syncthreads();

    // deterministic block reduction: 64 columns, sums then sumsq
    const int tid = threadIdx.x;
    if (tid < 64) {
        const int t = tid >> 4, c = tid & 15;
        float a0 = 0.f;
#pragma unroll
        for (int w = 0; w < 8; ++w)
            a0 += sS[w * 128 + t * 32 + c] + sS[w * 128 + t * 32 + 16 + c];
        partial[(size_t)blockIdx.x * 128 + tid] = a0;
    } else if (tid < 128) {
        const int j = tid - 64, t = j >> 4, c = j & 15;
        float a0 = 0.f;
#pragma unroll
        for (int w = 0; w < 8; ++w)
            a0 += sQ[w * 128 + t * 32 + c] + sQ[w * 128 + t * 32 + 16 + c];
        partial[(size_t)blockIdx.x * 128 + 64 + j] = a0;
    }
}

// ---------- kernel 3: reduce partials -> scale/shift per channel ----------
__global__ void bn_stats(const float* __restrict__ partial, int nblocks, float invN,
                         const float* __restrict__ gamma, const float* __restrict__ beta,
                         float* __restrict__ stats) {   // stats[0..63]=scale, [64..127]=shift
    __shared__ float s[128];
    const int j = threadIdx.x;  // 0..127
    float acc = 0.f;
    for (int b = 0; b < nblocks; ++b) acc += partial[(size_t)b * 128 + j];
    s[j] = acc;
    __syncthreads();
    if (j < 64) {
        const float mean = s[j] * invN;
        const float var  = s[64 + j] * invN - mean * mean;
        const float sc   = gamma[j] * rsqrtf(var + BN_EPS);
        stats[j]      = sc;
        stats[64 + j] = beta[j] - mean * sc;
    }
}

// ---------- kernel 4: in-place normalize + ReLU (float4) ----------
__global__ void bn_apply(float* __restrict__ y, const float* __restrict__ stats, int total4) {
    int i = blockIdx.x * blockDim.x + threadIdx.x;
    if (i < total4) {
        const int d = (i * 4) & (C_OUT - 1);   // 4 consecutive channels, no wrap
        float4 v = reinterpret_cast<float4*>(y)[i];
        v.x = v.x * stats[d + 0] + stats[64 + d + 0];
        v.y = v.y * stats[d + 1] + stats[64 + d + 1];
        v.z = v.z * stats[d + 2] + stats[64 + d + 2];
        v.w = v.w * stats[d + 3] + stats[64 + d + 3];
        v.x = v.x > 0.f ? v.x : 0.f;
        v.y = v.y > 0.f ? v.y : 0.f;
        v.z = v.z > 0.f ? v.z : 0.f;
        v.w = v.w > 0.f ? v.w : 0.f;
        reinterpret_cast<float4*>(y)[i] = v;
    }
}

// ---------- launch ----------
extern "C" void kernel_launch(void* const* d_in, const int* in_sizes, int n_in,
                              void* d_out, int out_size, void* d_ws, size_t ws_size,
                              hipStream_t stream) {
    const float* x     = (const float*)d_in[0];
    const float* W     = (const float*)d_in[1];
    // d_in[2] = bias b: cancels exactly under BN mean subtraction -> unused
    const float* gamma = (const float*)d_in[3];
    const float* beta  = (const float*)d_in[4];
    const int*   idx   = (const int*)d_in[5];

    const int N       = in_sizes[0] / C_IN;      // 262144
    const int nblocks = N / 256;                 // rows per block = 256 (8 waves x 32)

    // workspace layout
    char* ws = (char*)d_ws;
    unsigned short* xb      = (unsigned short*)(ws);                        // N*32 bf16
    unsigned short* wpacked = (unsigned short*)(ws + (size_t)N * C_IN * 2); // 55296 bf16
    float* partial = (float*)(ws + (size_t)N * C_IN * 2 + 131072);          // nblocks*128 f32
    float* stats   = (float*)((char*)partial + (size_t)nblocks * 128 * 4);  // 128 f32

    float* y = (float*)d_out;   // pre-BN y written here, normalized in place

    // 1a: x -> bf16
    {
        int total4 = (N * C_IN) / 4;
        cvt_x_bf16<<<(total4 + 255) / 256, 256, 0, stream>>>(x, xb, total4);
    }
    // 1b: pack W into B-fragment layout
    {
        int total = KVOL * 4 * 32 * 16;
        pack_w<<<(total + 255) / 256, 256, 0, stream>>>(W, wpacked);
    }
    // 2: gather + WMMA GEMM + partial BN stats
    sparse_conv_gemm<<<nblocks, 256, 0, stream>>>(xb, wpacked, idx, y, partial);
    // 3: finalize BN scale/shift
    bn_stats<<<1, 128, 0, stream>>>(partial, nblocks, 1.0f / (float)N, gamma, beta, stats);
    // 4: normalize + ReLU in place
    {
        int total4 = (N * C_OUT) / 4;
        bn_apply<<<(total4 + 255) / 256, 256, 0, stream>>>(y, stats, total4);
    }
}
